// TaxFraudHGT_72679436583024
// MI455X (gfx1250) — compile-verified, matrix operand order checked by reading, not run on hardware
//
#include <hip/hip_runtime.h>
#include <hip/hip_bf16.h>
#include <math.h>

// ---------------------------------------------------------------------------
// Problem constants (from reference)
// ---------------------------------------------------------------------------
#define NCOMP 200000
#define NOFF  20000
#define NPERS 100000
#define FEAT  15
#define HID   128
#define HEADS 8
#define DHEAD 16
#define LAYERS 2

typedef __attribute__((ext_vector_type(16))) __bf16 v16bf;
typedef __attribute__((ext_vector_type(8)))  __bf16 v8bf;
typedef __attribute__((ext_vector_type(8)))  float  v8f;

union Frag16 { v16bf v; struct { v8bf lo, hi; } halves; };

// ---------------------------------------------------------------------------
// Input projection: h = x @ W_in[t] + b_in[t]   (K=15, scalar f32)
// one block (128 threads) per node
// ---------------------------------------------------------------------------
__global__ __launch_bounds__(128) void hgt_input_proj(
    const float* __restrict__ x, const float* __restrict__ W,
    const float* __restrict__ b, float* __restrict__ h,
    __bf16* __restrict__ hb)
{
  __shared__ float xs[FEAT];
  const int n = blockIdx.x;
  const int j = threadIdx.x;
  if (j < FEAT) xs[j] = x[(size_t)n * FEAT + j];
  __syncthreads();
  float acc = b[j];
#pragma unroll
  for (int f = 0; f < FEAT; ++f) acc = fmaf(xs[f], W[f * HID + j], acc);
  h[(size_t)n * HID + j] = acc;
  if (hb) hb[(size_t)n * HID + j] = (__bf16)acc;
}

// ---------------------------------------------------------------------------
// Transpose-convert f32 [K][N] -> bf16 [N][K]  (batch over blockIdx.z)
// Makes the GEMM B operand (a column of W) contiguous in K.
// ---------------------------------------------------------------------------
__global__ void hgt_transpose_bf16(const float* __restrict__ src,
                                   __bf16* __restrict__ dst, int K, int N)
{
  const size_t bat = blockIdx.z;
  const float* s = src + bat * (size_t)K * N;
  __bf16*      d = dst + bat * (size_t)K * N;
  int i = blockIdx.x * blockDim.x + threadIdx.x;
  if (i >= K * N) return;
  int k = i / N, n = i % N;
  d[(size_t)n * K + k] = (__bf16)s[i];
}

// ---------------------------------------------------------------------------
// WMMA fragment loader contract (documented 16-bit A 16x32 layout, B mirrored):
//   lane L: row = L&15, kh = (L>>4)*8
//   halves.lo = 8 contiguous bf16 at [row, k0+kh]
//   halves.hi = 8 contiguous bf16 at [row, k0+16+kh]
// ---------------------------------------------------------------------------

// Fused kqv GEMM: C[M,384] = A[M,128] @ W[128,384] + b, split-stored to
// k (bf16), q (f32, nullable), v (bf16). One wave computes a 16x64 tile.
__global__ __launch_bounds__(32) void hgt_kqv_gemm(
    const __bf16* __restrict__ A,   // [M][128] bf16
    const __bf16* __restrict__ Bt,  // [384][128] bf16 (W column-contiguous)
    const float*  __restrict__ bias,// [384]
    __bf16* __restrict__ k_out, float* __restrict__ q_out,
    __bf16* __restrict__ v_out, int M)
{
  const int lane = threadIdx.x & 31;
  const int m0 = blockIdx.x * 16;
  const int n0 = blockIdx.y * 64;
  const int mr = lane & 15;
  const int kh = (lane >> 4) * 8;
  v8f acc[4] = {};
  const __bf16* arow = A + (size_t)(m0 + mr) * HID;
#pragma unroll
  for (int ks = 0; ks < HID; ks += 32) {
    Frag16 a;
    a.halves.lo = *(const v8bf*)(arow + ks + kh);
    a.halves.hi = *(const v8bf*)(arow + ks + 16 + kh);
#pragma unroll
    for (int t = 0; t < 4; ++t) {
      const __bf16* brow = Bt + (size_t)(n0 + t * 16 + mr) * HID;
      Frag16 b;
      b.halves.lo = *(const v8bf*)(brow + ks + kh);
      b.halves.hi = *(const v8bf*)(brow + ks + 16 + kh);
      acc[t] = __builtin_amdgcn_wmma_f32_16x16x32_bf16(
          false, a.v, false, b.v, (short)0, acc[t], false, false);
    }
  }
  const int rb = (lane >> 4) * 8;
#pragma unroll
  for (int t = 0; t < 4; ++t) {
    const int ncol = n0 + t * 16 + mr;
    const float bv = bias[ncol];
#pragma unroll
    for (int r = 0; r < 8; ++r) {
      const int row = m0 + rb + r;
      const float val = acc[t][r] + bv;
      if (ncol < HID) {
        k_out[(size_t)row * HID + ncol] = (__bf16)val;
      } else if (ncol < 2 * HID) {
        if (q_out) q_out[(size_t)row * HID + (ncol - HID)] = val;
      } else {
        v_out[(size_t)row * HID + (ncol - 2 * HID)] = (__bf16)val;
      }
    }
  }
}

// Generic WMMA GEMM: C[M,N] = A[M,128] @ W[128,N] + b (f32 out). grid.y = N/64.
__global__ __launch_bounds__(32) void hgt_gemm_bf16(
    const __bf16* __restrict__ A, const __bf16* __restrict__ Bt,
    const float* __restrict__ bias, float* __restrict__ C, int N)
{
  const int lane = threadIdx.x & 31;
  const int m0 = blockIdx.x * 16;
  const int n0 = blockIdx.y * 64;
  const int mr = lane & 15;
  const int kh = (lane >> 4) * 8;
  v8f acc[4] = {};
  const __bf16* arow = A + (size_t)(m0 + mr) * HID;
#pragma unroll
  for (int ks = 0; ks < HID; ks += 32) {
    Frag16 a;
    a.halves.lo = *(const v8bf*)(arow + ks + kh);
    a.halves.hi = *(const v8bf*)(arow + ks + 16 + kh);
#pragma unroll
    for (int t = 0; t < 4; ++t) {
      const __bf16* brow = Bt + (size_t)(n0 + t * 16 + mr) * HID;
      Frag16 b;
      b.halves.lo = *(const v8bf*)(brow + ks + kh);
      b.halves.hi = *(const v8bf*)(brow + ks + 16 + kh);
      acc[t] = __builtin_amdgcn_wmma_f32_16x16x32_bf16(
          false, a.v, false, b.v, (short)0, acc[t], false, false);
    }
  }
  const int rb = (lane >> 4) * 8;
#pragma unroll
  for (int t = 0; t < 4; ++t) {
    const int ncol = n0 + t * 16 + mr;
    const float bv = bias[ncol];
#pragma unroll
    for (int r = 0; r < 8; ++r)
      C[(size_t)(m0 + rb + r) * N + ncol] = acc[t][r] + bv;
  }
}

// Per-head relation transform: k_e[:,h,:] = k[:,h,:] @ a_rel[h] (16x16),
// same for v with m_rel. K=16 zero-padded to 32. One wave per (16 nodes, head).
__global__ __launch_bounds__(32) void hgt_rel_transform(
    const __bf16* __restrict__ kin, const __bf16* __restrict__ vin,
    const __bf16* __restrict__ aT,  const __bf16* __restrict__ mT, // [8][16][16]
    __bf16* __restrict__ keo, __bf16* __restrict__ veo, int M)
{
  const int lane = threadIdx.x & 31;
  const int m0 = blockIdx.x * 16;
  const int hh = blockIdx.y;
  const int mr = lane & 15;
  const int kh = (lane >> 4) * 8;
  v8bf zero;
#pragma unroll
  for (int i = 0; i < 8; ++i) zero[i] = (__bf16)0.0f;

  Frag16 a, b;
  a.halves.hi = zero;
  b.halves.hi = zero;
  a.halves.lo = *(const v8bf*)(kin + (size_t)(m0 + mr) * HID + hh * DHEAD + kh);
  b.halves.lo = *(const v8bf*)(aT + ((size_t)hh * DHEAD + mr) * DHEAD + kh);
  v8f c = {};
  c = __builtin_amdgcn_wmma_f32_16x16x32_bf16(false, a.v, false, b.v, (short)0,
                                              c, false, false);
  const int rb = (lane >> 4) * 8;
#pragma unroll
  for (int r = 0; r < 8; ++r)
    keo[(size_t)(m0 + rb + r) * HID + hh * DHEAD + mr] = (__bf16)c[r];

  a.halves.lo = *(const v8bf*)(vin + (size_t)(m0 + mr) * HID + hh * DHEAD + kh);
  b.halves.lo = *(const v8bf*)(mT + ((size_t)hh * DHEAD + mr) * DHEAD + kh);
  v8f c2 = {};
  c2 = __builtin_amdgcn_wmma_f32_16x16x32_bf16(false, a.v, false, b.v, (short)0,
                                               c2, false, false);
#pragma unroll
  for (int r = 0; r < 8; ++r)
    veo[(size_t)(m0 + rb + r) * HID + hh * DHEAD + mr] = (__bf16)c2[r];
}

// ---------------------------------------------------------------------------
// Edge phase. Softmax is shift-invariant, so we skip the segment-max pass:
// ex = exp(s); denom = segsum(ex); alpha = ex/(denom+1e-16).
// ---------------------------------------------------------------------------
__global__ void hgt_score(const float* __restrict__ q, const __bf16* __restrict__ ke,
                          const int* __restrict__ ei, int E,
                          const float* __restrict__ p_rel,   // [8] for (l,e)
                          float* __restrict__ exs,           // [E][8] pre-offset
                          float* __restrict__ denom)         // [N_C][8]
{
  int t = blockIdx.x * blockDim.x + threadIdx.x;
  if (t >= E * HEADS) return;
  int e = t >> 3, h = t & 7;
  int src = ei[e], dst = ei[E + e];
  const float*  qp = q  + (size_t)dst * HID + h * DHEAD;
  const __bf16* kp = ke + (size_t)src * HID + h * DHEAD;
  float s = 0.f;
#pragma unroll
  for (int d = 0; d < DHEAD; ++d) s = fmaf(qp[d], (float)kp[d], s);
  s *= p_rel[h] * 0.25f;           // p_rel / sqrt(D), D=16
  float ex = __expf(s);
  exs[t] = ex;
  atomicAdd(&denom[(size_t)dst * HEADS + h], ex);
}

__global__ void hgt_aggregate(const __bf16* __restrict__ ve, const int* __restrict__ ei,
                              int E, const float* __restrict__ exs,
                              const float* __restrict__ denom, float* __restrict__ agg)
{
  int t = blockIdx.x * blockDim.x + threadIdx.x;
  if (t >= E * HEADS) return;
  int e = t >> 3, h = t & 7;
  int src = ei[e], dst = ei[E + e];
  float alpha = exs[t] / (denom[(size_t)dst * HEADS + h] + 1e-16f);
  const __bf16* vp = ve + (size_t)src * HID + h * DHEAD;
  float* ap = agg + (size_t)dst * HID + h * DHEAD;
#pragma unroll
  for (int d = 0; d < DHEAD; ++d) atomicAdd(&ap[d], alpha * (float)vp[d]);
}

// exact gelu -> bf16
__global__ void hgt_gelu(const float* __restrict__ in, __bf16* __restrict__ out, int n)
{
  int i = blockIdx.x * blockDim.x + threadIdx.x;
  if (i >= n) return;
  float x = in[i];
  out[i] = (__bf16)(0.5f * x * (1.f + erff(x * 0.70710678118654752f)));
}

// skip-gate + ELU, in-place update of company features (f32 + bf16 copies)
__global__ void hgt_epilogue(const float* __restrict__ outc,
                             const float* __restrict__ skip_ptr,
                             float* __restrict__ h_c, __bf16* __restrict__ hb_c, int n)
{
  int i = blockIdx.x * blockDim.x + threadIdx.x;
  if (i >= n) return;
  float a = 1.f / (1.f + __expf(-skip_ptr[0]));
  float hn = a * outc[i] + (1.f - a) * h_c[i];
  float el = hn > 0.f ? hn : expm1f(hn);
  h_c[i] = el;
  hb_c[i] = (__bf16)el;
}

// classifier head: relu(h@W1+b1)@W2+b2 ; one wave (32 lanes = 32 hidden) per node
__global__ __launch_bounds__(256) void hgt_classifier(
    const float* __restrict__ h, const float* __restrict__ W1,
    const float* __restrict__ b1, const float* __restrict__ W2,
    const float* __restrict__ b2, float* __restrict__ out, int N)
{
  const int wave = threadIdx.x >> 5;
  const int lane = threadIdx.x & 31;
  const int n = blockIdx.x * 8 + wave;
  if (n >= N) return;
  const float* hp = h + (size_t)n * HID;
  float acc = b1[lane];
#pragma unroll 4
  for (int k = 0; k < HID; ++k) acc = fmaf(hp[k], W1[k * 32 + lane], acc);
  acc = fmaxf(acc, 0.f);
  float pr = acc * W2[lane];
#pragma unroll
  for (int off = 16; off > 0; off >>= 1) pr += __shfl_down(pr, off, 32);
  if (lane == 0) out[n] = pr + b2[0];
}

// ---------------------------------------------------------------------------
// Launch
// ---------------------------------------------------------------------------
extern "C" void kernel_launch(void* const* d_in, const int* in_sizes, int n_in,
                              void* d_out, int out_size, void* d_ws, size_t ws_size,
                              hipStream_t stream)
{
  const float* x_c   = (const float*)d_in[0];
  const float* x_o   = (const float*)d_in[1];
  const float* x_p   = (const float*)d_in[2];
  const float* W_in  = (const float*)d_in[3];
  const float* b_in  = (const float*)d_in[4];
  const float* kqv_W = (const float*)d_in[5];
  const float* kqv_b = (const float*)d_in[6];
  const float* a_rel = (const float*)d_in[7];
  const float* m_rel = (const float*)d_in[8];
  const float* p_rel = (const float*)d_in[9];
  const float* out_W = (const float*)d_in[10];
  const float* out_b = (const float*)d_in[11];
  const float* skip  = (const float*)d_in[12];
  const float* clfW1 = (const float*)d_in[13];
  const float* clfb1 = (const float*)d_in[14];
  const float* clfW2 = (const float*)d_in[15];
  const float* clfb2 = (const float*)d_in[16];
  const int* ei[5] = {(const int*)d_in[17], (const int*)d_in[18],
                      (const int*)d_in[19], (const int*)d_in[20],
                      (const int*)d_in[21]};
  const int  ECNT[5] = {200000, 150000, 50000, 25000, 400000};
  const int  ESRC[5] = {0, 2, 0, 0, 0};   // company, person, company...
  float* out = (float*)d_out;

  // ---- carve workspace ----
  char* cur = (char*)d_ws;
  auto alloc = [&](size_t bytes) -> void* {
    void* r = (void*)cur;
    cur += (bytes + 255) & ~(size_t)255;
    return r;
  };
  float*  h_c  = (float*)alloc((size_t)NCOMP * HID * 4);
  float*  h_o  = (float*)alloc((size_t)NOFF  * HID * 4);
  float*  h_p  = (float*)alloc((size_t)NPERS * HID * 4);
  __bf16* hb_c = (__bf16*)alloc((size_t)NCOMP * HID * 2);
  __bf16* hb_p = (__bf16*)alloc((size_t)NPERS * HID * 2);
  float*  q_c  = (float*)alloc((size_t)NCOMP * HID * 4);
  __bf16* k_c  = (__bf16*)alloc((size_t)NCOMP * HID * 2);
  __bf16* v_c  = (__bf16*)alloc((size_t)NCOMP * HID * 2);
  __bf16* k_p  = (__bf16*)alloc((size_t)NPERS * HID * 2);
  __bf16* v_p  = (__bf16*)alloc((size_t)NPERS * HID * 2);
  __bf16* ke[5]; __bf16* ve[5];
  for (int e = 0; e < 5; ++e) {
    size_t ns = (ESRC[e] == 0) ? NCOMP : NPERS;
    ke[e] = (__bf16*)alloc(ns * HID * 2);
    ve[e] = (__bf16*)alloc(ns * HID * 2);
  }
  float*  exs   = (float*)alloc((size_t)(200000 + 150000 + 50000 + 25000 + 400000) * HEADS * 4);
  float*  denom = (float*)alloc((size_t)NCOMP * HEADS * 4);
  float*  agg   = (float*)alloc((size_t)NCOMP * HID * 4);
  __bf16* gelub = (__bf16*)alloc((size_t)NCOMP * HID * 2);
  float*  outc  = (float*)alloc((size_t)NCOMP * HID * 4);
  __bf16* kqvT  = (__bf16*)alloc((size_t)LAYERS * 3 * 384 * 128 * 2);
  __bf16* outT  = (__bf16*)alloc((size_t)LAYERS * 128 * 128 * 2);
  __bf16* relTa = (__bf16*)alloc((size_t)LAYERS * 5 * 8 * 256 * 2);
  __bf16* relTm = (__bf16*)alloc((size_t)LAYERS * 5 * 8 * 256 * 2);

  // ---- weight conversion (re-done every call; deterministic) ----
  hgt_transpose_bf16<<<dim3((128 * 384 + 255) / 256, 1, LAYERS * 3), 256, 0, stream>>>(
      kqv_W, kqvT, 128, 384);
  hgt_transpose_bf16<<<dim3((128 * 128 + 255) / 256, 1, LAYERS), 256, 0, stream>>>(
      out_W, outT, 128, 128);
  hgt_transpose_bf16<<<dim3(1, 1, LAYERS * 5 * 8), 256, 0, stream>>>(a_rel, relTa, 16, 16);
  hgt_transpose_bf16<<<dim3(1, 1, LAYERS * 5 * 8), 256, 0, stream>>>(m_rel, relTm, 16, 16);

  // ---- input projections ----
  hgt_input_proj<<<NCOMP, 128, 0, stream>>>(x_c, W_in + 0 * FEAT * HID,
                                            b_in + 0 * HID, h_c, hb_c);
  hgt_input_proj<<<NOFF, 128, 0, stream>>>(x_o, W_in + 1 * FEAT * HID,
                                           b_in + 1 * HID, h_o, (__bf16*)nullptr);
  hgt_input_proj<<<NPERS, 128, 0, stream>>>(x_p, W_in + 2 * FEAT * HID,
                                            b_in + 2 * HID, h_p, hb_p);

  // ---- HGT layers ----
  for (int l = 0; l < LAYERS; ++l) {
    // kqv projections (company: k,q,v ; person: k,v — offshore never a source)
    hgt_kqv_gemm<<<dim3(NCOMP / 16, 6), 32, 0, stream>>>(
        hb_c, kqvT + (size_t)(l * 3 + 0) * 384 * 128, kqv_b + (l * 3 + 0) * 384,
        k_c, q_c, v_c, NCOMP);
    hgt_kqv_gemm<<<dim3(NPERS / 16, 6), 32, 0, stream>>>(
        hb_p, kqvT + (size_t)(l * 3 + 2) * 384 * 128, kqv_b + (l * 3 + 2) * 384,
        k_p, (float*)nullptr, v_p, NPERS);

    // per-edge-type relation transforms
    for (int e = 0; e < 5; ++e) {
      const __bf16* kin = (ESRC[e] == 0) ? k_c : k_p;
      const __bf16* vin = (ESRC[e] == 0) ? v_c : v_p;
      int M = (ESRC[e] == 0) ? NCOMP : NPERS;
      hgt_rel_transform<<<dim3(M / 16, HEADS), 32, 0, stream>>>(
          kin, vin, relTa + (size_t)(l * 5 + e) * 8 * 256,
          relTm + (size_t)(l * 5 + e) * 8 * 256, ke[e], ve[e], M);
    }

    hipMemsetAsync(denom, 0, (size_t)NCOMP * HEADS * 4, stream);
    hipMemsetAsync(agg, 0, (size_t)NCOMP * HID * 4, stream);

    // scores + exp + segment denom
    size_t eoff = 0;
    for (int e = 0; e < 5; ++e) {
      int total = ECNT[e] * HEADS;
      hgt_score<<<(total + 255) / 256, 256, 0, stream>>>(
          q_c, ke[e], ei[e], ECNT[e], p_rel + (l * 5 + e) * HEADS,
          exs + eoff * HEADS, denom);
      eoff += ECNT[e];
    }
    // normalize + scatter-add messages
    eoff = 0;
    for (int e = 0; e < 5; ++e) {
      int total = ECNT[e] * HEADS;
      hgt_aggregate<<<(total + 255) / 256, 256, 0, stream>>>(
          ve[e], ei[e], ECNT[e], exs + eoff * HEADS, denom, agg);
      eoff += ECNT[e];
    }

    // gelu -> out linear -> skip gate + elu
    hgt_gelu<<<(NCOMP * HID + 255) / 256, 256, 0, stream>>>(agg, gelub, NCOMP * HID);
    hgt_gemm_bf16<<<dim3(NCOMP / 16, 2), 32, 0, stream>>>(
        gelub, outT + (size_t)l * 128 * 128, out_b + l * HID, outc, HID);
    hgt_epilogue<<<(NCOMP * HID + 255) / 256, 256, 0, stream>>>(
        outc, skip + l, h_c, hb_c, NCOMP * HID);
  }

  // ---- classifier heads (company uses final h_c; others use input proj) ----
  hgt_classifier<<<(NCOMP + 7) / 8, 256, 0, stream>>>(
      h_c, clfW1 + 0 * HID * 32, clfb1 + 0 * 32, clfW2 + 0 * 32, clfb2 + 0,
      out + 0, NCOMP);
  hgt_classifier<<<(NOFF + 7) / 8, 256, 0, stream>>>(
      h_o, clfW1 + 1 * HID * 32, clfb1 + 1 * 32, clfW2 + 1 * 32, clfb2 + 1,
      out + NCOMP, NOFF);
  hgt_classifier<<<(NPERS + 7) / 8, 256, 0, stream>>>(
      h_p, clfW1 + 2 * HID * 32, clfb1 + 2 * 32, clfW2 + 2 * 32, clfb2 + 2,
      out + NCOMP + NOFF, NPERS);
}
